// Head_69131793597006
// MI455X (gfx1250) — matrix operand order repaired and checked by reading, not compiled, not run on hardware
//
#include <hip/hip_runtime.h>
#include <hip/hip_bf16.h>

// ---------------------------------------------------------------------------
// Causal single-head attention for MI455X (gfx1250), bf16 WMMA + fp32 softmax.
//   x[8,2048,1024] f32, W_{k,q,v}[1024,64] f32  ->  out[8,2048,64] f32
// ~11 GFLOP vs ~75MB HBM traffic -> memory-bound on the v_wmma bf16 path.
// This revision replaces the software bf16-round sequence (3 VALU/elt) with
// v_cvt_pk_bf16_f32 and restructures QKV so each wave owns a distinct M tile
// (conversions amortized over 12 WMMAs instead of 3).
// ---------------------------------------------------------------------------

typedef __attribute__((ext_vector_type(16))) __bf16 v16bf;
typedef __attribute__((ext_vector_type(2)))  __bf16 v2bf;
typedef __attribute__((ext_vector_type(2)))  float  v2f;
typedef __attribute__((ext_vector_type(8)))  float  v8f;

union FragAB {
  v16bf v;
  unsigned int d[8];
  unsigned short u[16];
};

// packed f32x2 -> bf16x2 (RNE); prefers the native v_cvt_pk_bf16_f32
__device__ __forceinline__ unsigned int pk_bf16(float lo, float hi) {
#if __has_builtin(__builtin_amdgcn_cvt_pk_bf16_f32)
  auto p = __builtin_amdgcn_cvt_pk_bf16_f32(lo, hi);
  return __builtin_bit_cast(unsigned int, p);
#else
  v2f f; f[0] = lo; f[1] = hi;
  v2bf p = __builtin_convertvector(f, v2bf);
  return __builtin_bit_cast(unsigned int, p);
#endif
}

__device__ __forceinline__ unsigned short bf1(float f) {
  __bf16 h = (__bf16)f;
  return __builtin_bit_cast(unsigned short, h);
}

// ---------------------------------------------------------------------------
// Kernel 0: weights [1024,64] f32 -> wT [3][64][1024] bf16 (transposed) so
// WMMA B-fragments are contiguous bf16-pair (dword) loads.
// Section order: 0 = W_q, 1 = W_k, 2 = W_v.
// ---------------------------------------------------------------------------
__global__ void __launch_bounds__(256) wt_kernel(
    const float* __restrict__ Wq, const float* __restrict__ Wk,
    const float* __restrict__ Wv, unsigned short* __restrict__ wT) {
  int tid = blockIdx.x * blockDim.x + threadIdx.x;   // < 3*65536
  int mat = tid >> 16;                               // uniform per block
  int rem = tid & 65535;
  int c = rem >> 10;          // output column (0..63)
  int k = rem & 1023;         // reduction index (0..1023)
  const float* W = (mat == 0) ? Wq : (mat == 1) ? Wk : Wv;
  wT[tid] = bf1(W[k * 64 + c]);
}

// ---------------------------------------------------------------------------
// Kernel 1: QKV projection. Block = 128 threads (4 waves); block covers 64
// token rows, wave w owns rows [w*16, w*16+16) and computes ALL 12 output
// 16x16 tiles (Q0..3, K0..3, V0..3) -> A fragment converted once per k-step.
//   qws, kws : bf16 [B*T, 64] row-major
//   vT       : bf16 [B, 64, T]  (transposed -> contiguous B-frags in phase 2)
// ---------------------------------------------------------------------------
__global__ void __launch_bounds__(128) qkv_kernel(
    const float* __restrict__ x, const unsigned short* __restrict__ wT,
    unsigned short* __restrict__ qws, unsigned short* __restrict__ kws,
    unsigned short* __restrict__ vT) {
  const int tid  = threadIdx.x;
  const int lane = tid & 31;
  const int w    = tid >> 5;
  const int hlf  = lane >> 4;
  const int ln   = lane & 15;
  const int rowbase = blockIdx.x * 64 + w * 16;   // global token row

  v8f acc[12];
#pragma unroll
  for (int t = 0; t < 12; ++t) acc[t] = (v8f){};

  const float* xrow = x + (size_t)(rowbase + ln) * 1024;

  for (int k0 = 0; k0 < 1024; k0 += 32) {
    // A fragment (16x32 bf16): lane row = ln; K = half*16 + 2v (+0,1)
    FragAB a;
#pragma unroll
    for (int v = 0; v < 8; ++v) {
      int k = (hlf << 4) + (v << 1);
      float2 f = *(const float2*)(xrow + k0 + k);
      a.d[v] = pk_bf16(f.x, f.y);
    }
    // 12 B fragments (32x16 bf16): lane col; K = half*16 + 2v (+0,1)
#pragma unroll
    for (int t = 0; t < 12; ++t) {
      const unsigned short* wp =
          wT + (t >> 2) * 65536 + (size_t)((t & 3) * 16 + ln) * 1024 + k0;
      FragAB b;
#pragma unroll
      for (int v = 0; v < 8; ++v)
        b.d[v] = *(const unsigned int*)(wp + (hlf << 4) + (v << 1));
      acc[t] = __builtin_amdgcn_wmma_f32_16x16x32_bf16(
          false, a.v, false, b.v, (short)0, acc[t], false, false);
    }
  }

  // D layout: VGPR r holds (M = r + 8*half, N = ln)
  const int bb   = rowbase >> 11;      // batch (tile never crosses batch)
  const int tloc = rowbase & 2047;     // token within batch
#pragma unroll
  for (int r = 0; r < 8; ++r) {
    int m = r + (hlf << 3);
    size_t grow = (size_t)(rowbase + m) * 64;
#pragma unroll
    for (int c = 0; c < 4; ++c) {
      qws[grow + c * 16 + ln] = bf1(acc[c][r]);
      kws[grow + c * 16 + ln] = bf1(acc[4 + c][r]);
      vT[(size_t)(bb * 64 + c * 16 + ln) * 2048 + tloc + m] = bf1(acc[8 + c][r]);
    }
  }
}

// ---------------------------------------------------------------------------
// Kernel 2: flash attention. Block = (batch, 64-query block), 128 threads.
// Each of 4 waves owns a 16-row query stripe: no cross-wave synchronization.
// Per 64-key block: 8 WMMAs for S=Q*K^T, online softmax (cross-lane reductions
// within 16-lane halves matching the C/D row layout), P staged to per-wave LDS
// as bf16 (s_wait_dscnt hazard), then 8 WMMAs for O += P*V (alpha-rescaled
// accumulator passed as the WMMA C operand).
// ---------------------------------------------------------------------------
__global__ void __launch_bounds__(128) attn_kernel(
    const unsigned short* __restrict__ qws, const unsigned short* __restrict__ kws,
    const unsigned short* __restrict__ vT, float* __restrict__ out) {
  __shared__ unsigned short Plds[4][16][64];   // per-wave P tile (8 KB)

  const int tid  = threadIdx.x;
  const int lane = tid & 31;
  const int w    = tid >> 5;
  const int hlf  = lane >> 4;
  const int ln   = lane & 15;
  const int bb   = blockIdx.x >> 5;     // batch
  const int qb   = blockIdx.x & 31;     // 64-row query block
  const int q0   = qb * 64 + w * 16;    // query row base within batch
  const size_t tokbase = (size_t)bb * 2048;
  const float scale = 0.03125f;         // 1024^-0.5

  // Q A-fragments for h in [0,32) and [32,64), kept in registers
  FragAB qa[2];
  {
    const unsigned short* qrow = qws + (tokbase + q0 + ln) * 64;
#pragma unroll
    for (int hs = 0; hs < 2; ++hs)
#pragma unroll
      for (int v = 0; v < 8; ++v) {
        int k = hs * 32 + (hlf << 4) + (v << 1);
        qa[hs].d[v] = *(const unsigned int*)(qrow + k);
      }
  }

  v8f O0 = {}, O1 = {}, O2 = {}, O3 = {};
  float mrow[8], lrow[8];
#pragma unroll
  for (int r = 0; r < 8; ++r) { mrow[r] = -INFINITY; lrow[r] = 0.0f; }

  for (int kb = 0; kb <= qb; ++kb) {
    // ---- S = Q * K^T (4 key sub-tiles of 16, contraction over H=64) ----
    v8f S[4];
#pragma unroll
    for (int t = 0; t < 4; ++t) {
      v8f a = {};
      const unsigned short* krow = kws + (tokbase + kb * 64 + t * 16 + ln) * 64;
#pragma unroll
      for (int hs = 0; hs < 2; ++hs) {
        FragAB kf;
#pragma unroll
        for (int v = 0; v < 8; ++v)
          kf.d[v] = *(const unsigned int*)(krow + hs * 32 + hlf * 16 + 2 * v);
        a = __builtin_amdgcn_wmma_f32_16x16x32_bf16(false, qa[hs].v, false, kf.v, (short)0, a, false, false);
      }
      S[t] = a;
    }

    // ---- online softmax (row = (r, half) spans 16 lanes of one half) ----
#pragma unroll
    for (int r = 0; r < 8; ++r) {
      int m = r + (hlf << 3);
      int qidx = q0 + m;
      float sv[4];
      float rmax = -INFINITY;
#pragma unroll
      for (int t = 0; t < 4; ++t) {
        int key = kb * 64 + t * 16 + ln;
        float s = S[t][r] * scale;
        s = (key > qidx) ? -INFINITY : s;    // causal mask (select, no branch)
        sv[t] = s;
        rmax = fmaxf(rmax, s);
      }
#pragma unroll
      for (int off = 8; off >= 1; off >>= 1)
        rmax = fmaxf(rmax, __shfl_xor(rmax, off, 16));
      float mnew  = fmaxf(mrow[r], rmax);    // finite: diag block has key<=q
      float alpha = __expf(mrow[r] - mnew);
      mrow[r] = mnew;
      float rs = 0.0f;
#pragma unroll
      for (int t = 0; t < 4; ++t) {
        float p = __expf(sv[t] - mnew);
        rs += p;
        Plds[w][m][t * 16 + ln] = bf1(p);
      }
#pragma unroll
      for (int off = 8; off >= 1; off >>= 1)
        rs += __shfl_xor(rs, off, 16);
      lrow[r] = lrow[r] * alpha + rs;
      O0[r] *= alpha; O1[r] *= alpha; O2[r] *= alpha; O3[r] *= alpha;
    }

    // P stores -> A-fragment loads: same wave, explicit DS counter drain
    asm volatile("s_wait_dscnt 0" ::: "memory");

    // ---- O += P * V (contraction over 64 keys in two 32-steps) ----
#pragma unroll
    for (int ss = 0; ss < 2; ++ss) {
      FragAB pa;
      const unsigned int* prow = (const unsigned int*)&Plds[w][ln][0];
#pragma unroll
      for (int v = 0; v < 8; ++v) {
        int k = ss * 32 + (hlf << 4) + (v << 1);
        pa.d[v] = prow[k >> 1];
      }
#pragma unroll
      for (int th = 0; th < 4; ++th) {
        FragAB vf;
        const unsigned short* vrow =
            vT + (size_t)(bb * 64 + th * 16 + ln) * 2048 + kb * 64;
#pragma unroll
        for (int v = 0; v < 8; ++v)
          vf.d[v] = *(const unsigned int*)(vrow + ss * 32 + hlf * 16 + 2 * v);
        v8f& Ot = (th == 0) ? O0 : (th == 1) ? O1 : (th == 2) ? O2 : O3;
        Ot = __builtin_amdgcn_wmma_f32_16x16x32_bf16(false, pa.v, false, vf.v, (short)0, Ot, false, false);
      }
    }
  }

  // ---- epilogue: O / l, write f32 out [B,T,64] ----
#pragma unroll
  for (int r = 0; r < 8; ++r) {
    int m = r + (hlf << 3);
    float inv = 1.0f / lrow[r];
    float* orow = out + (tokbase + q0 + m) * 64;
    orow[ 0 + ln] = O0[r] * inv;
    orow[16 + ln] = O1[r] * inv;
    orow[32 + ln] = O2[r] * inv;
    orow[48 + ln] = O3[r] * inv;
  }
}

// ---------------------------------------------------------------------------
// Host launcher. Workspace layout (bf16 as ushort):
//   wT  [3*64*1024]      = 384 KB
//   qws [16384*64]       = 2 MB
//   kws [16384*64]       = 2 MB
//   vT  [8*64*2048]      = 2 MB        total ~6.4 MB
// ---------------------------------------------------------------------------
extern "C" void kernel_launch(void* const* d_in, const int* in_sizes, int n_in,
                              void* d_out, int out_size, void* d_ws, size_t ws_size,
                              hipStream_t stream) {
  const float* x  = (const float*)d_in[0];
  const float* Wk = (const float*)d_in[1];
  const float* Wq = (const float*)d_in[2];
  const float* Wv = (const float*)d_in[3];

  unsigned short* wT  = (unsigned short*)d_ws;
  unsigned short* qws = wT + 3 * 64 * 1024;
  unsigned short* kws = qws + 16384 * 64;
  unsigned short* vTt = kws + 16384 * 64;
  float* out = (float*)d_out;

  wt_kernel<<<dim3(768), dim3(256), 0, stream>>>(Wq, Wk, Wv, wT);
  qkv_kernel<<<dim3(256), dim3(128), 0, stream>>>(x, wT, qws, kws, vTt);
  attn_kernel<<<dim3(256), dim3(128), 0, stream>>>(qws, kws, vTt, out);
}
